// MultiHeadSelfAttention_61263413510415
// MI455X (gfx1250) — compile-verified, matrix operand order checked by reading, not run on hardware
//
#include <hip/hip_runtime.h>
#include <hip/hip_bf16.h>

// ---------------------------------------------------------------------------
// MHA for MI455X (gfx1250): bf16 WMMA GEMMs with async global->LDS staging
// + flash-attention (online softmax).
// ---------------------------------------------------------------------------

typedef __attribute__((ext_vector_type(8)))  __bf16 bf16x8;
typedef __attribute__((ext_vector_type(16))) __bf16 bf16x16;
typedef __attribute__((ext_vector_type(8)))  float  f32x8;

#define D_MODEL 768
#define SEQ     2048
#define BATCH   2
#define NHEAD   12
#define DKH     64
#define TOKENS  (BATCH * SEQ)   // 4096

// round-to-nearest-even f32 -> bf16
static __device__ __forceinline__ __bf16 f2bf(float f) {
  unsigned u = __float_as_uint(f);
  u += 0x7FFFu + ((u >> 16) & 1u);
  union { unsigned short s; __bf16 b; } cv;
  cv.s = (unsigned short)(u >> 16);
  return cv.b;
}

// Build a 16-wide bf16 fragment from two 16-byte (b128) loads.
// CDNA5 16-bit A/B fragment: elems 0..7 = K(half*8 + 0..7), elems 8..15 = K(16 + half*8 + 0..7)
static __device__ __forceinline__ bf16x16 ldfrag(const __bf16* p0, const __bf16* p1) {
  bf16x8 lo = *(const bf16x8*)p0;
  bf16x8 hi = *(const bf16x8*)p1;
  return __builtin_shufflevector(lo, hi, 0,1,2,3,4,5,6,7,8,9,10,11,12,13,14,15);
}

static __device__ __forceinline__ f32x8 wmma_bf16(bf16x16 a, bf16x16 b, f32x8 c) {
  return __builtin_amdgcn_wmma_f32_16x16x32_bf16(false, a, false, b, (short)0, c, false, false);
}

// Async copy of 16 bytes global -> LDS (CDNA5 GLOBAL_LOAD_ASYNC_TO_LDS_B128).
// lds_off: LDS byte address (low 32 bits of the generic pointer — flat
// aperture address).  Tracked by ASYNCcnt.
static __device__ __forceinline__ void async_cp16(unsigned lds_off, const void* g) {
  asm volatile("global_load_async_to_lds_b128 %0, %1, off"
               :: "v"(lds_off), "v"((unsigned long long)(uintptr_t)g)
               : "memory");
}
static __device__ __forceinline__ void wait_async0() {
  asm volatile("s_wait_asynccnt 0x0" ::: "memory");
}

// ---------------------------------------------------------------------------
__global__ void cvt_f32_bf16(const float* __restrict__ src, __bf16* __restrict__ dst, int n) {
  int i = blockIdx.x * blockDim.x + threadIdx.x;
  if (i < n) dst[i] = f2bf(src[i]);
}

// ---------------------------------------------------------------------------
// C[M,N] = A[M,K] * W[N,K]^T   (both operands K-contiguous, bf16, f32 accum)
// MODE 0: store bf16 row-major [M,N]
// MODE 1: store bf16 transposed [N, ldT]  (ldT = M)
// MODE 2: store f32 row-major [M,N]
//
// Block = 256 threads (8 waves, 2x4 wave grid) computing a 128x256 C tile;
// each wave owns a 64x64 sub-tile (4x4 grid of 16x16 WMMA accumulators).
// A (128x32) and B (256x32) bf16 k-slices are staged into double-buffered LDS
// via async global->LDS b128 copies, overlapping DMA with WMMA.
template<int MODE>
__global__ __launch_bounds__(256)
void gemm_bf16_nt(const __bf16* __restrict__ A, const __bf16* __restrict__ W,
                  void* __restrict__ out, int M, int N, int K, int ldT) {
  __shared__ __bf16 Abuf[2][128 * 32];   // 2 x 8 KB
  __shared__ __bf16 Bbuf[2][256 * 32];   // 2 x 16 KB

  const int tid  = threadIdx.x;
  const int lane = tid & 31;
  const int wave = tid >> 5;
  const int tilesNb = N >> 8;                       // N / 256
  const int bM = blockIdx.x / tilesNb, bN = blockIdx.x % tilesNb;
  const int mbBlk = bM << 7, nbBlk = bN << 8;
  const int wm = wave >> 2, wn = wave & 3;          // 2x4 wave grid
  const int mrow = lane & 15, half = lane >> 4, h8 = half * 8;

  // 16B-chunk assignments for async staging:
  // A tile = 512 chunks (2/thread), B tile = 1024 chunks (4/thread)
  const int crow = tid >> 2;                        // 0..63
  const int cb   = (tid & 3) * 16;                  // byte offset in 64B row
  const char* agp[2]; unsigned alo[2];
  const char* bgp[4]; unsigned blo[4];
#pragma unroll
  for (int j = 0; j < 2; ++j) {
    const int row = crow + 64 * j;
    agp[j] = (const char*)(A + (size_t)(mbBlk + row) * K) + cb;
    alo[j] = (unsigned)(row * 64 + cb);
  }
#pragma unroll
  for (int j = 0; j < 4; ++j) {
    const int row = crow + 64 * j;
    bgp[j] = (const char*)(W + (size_t)(nbBlk + row) * K) + cb;
    blo[j] = (unsigned)(row * 64 + cb);
  }
  const unsigned aBase[2] = { (unsigned)(uintptr_t)&Abuf[0][0],
                              (unsigned)(uintptr_t)&Abuf[1][0] };
  const unsigned bBase[2] = { (unsigned)(uintptr_t)&Bbuf[0][0],
                              (unsigned)(uintptr_t)&Bbuf[1][0] };

  f32x8 acc[4][4] = {};

  // prologue: stage k-tile 0 into buffer 0
#pragma unroll
  for (int j = 0; j < 2; ++j) async_cp16(aBase[0] + alo[j], agp[j]);
#pragma unroll
  for (int j = 0; j < 4; ++j) async_cp16(bBase[0] + blo[j], bgp[j]);
  wait_async0();
  __syncthreads();

  for (int kk = 0; kk < K; kk += 32) {
    const int cur = (kk >> 5) & 1;
    if (kk + 32 < K) {                               // stage next k-slice
      const int nxt = cur ^ 1;
      const int kb  = (kk + 32) * 2;                 // byte offset
#pragma unroll
      for (int j = 0; j < 2; ++j) async_cp16(aBase[nxt] + alo[j], agp[j] + kb);
#pragma unroll
      for (int j = 0; j < 4; ++j) async_cp16(bBase[nxt] + blo[j], bgp[j] + kb);
    }

    bf16x16 af[4], bfr[4];
#pragma unroll
    for (int i = 0; i < 4; ++i) {
      const __bf16* ap = &Abuf[cur][(wm * 64 + i * 16 + mrow) * 32 + h8];
      const __bf16* bp = &Bbuf[cur][(wn * 64 + i * 16 + mrow) * 32 + h8];
      af[i]  = ldfrag(ap, ap + 16);
      bfr[i] = ldfrag(bp, bp + 16);
    }
#pragma unroll
    for (int mi = 0; mi < 4; ++mi)
#pragma unroll
      for (int ni = 0; ni < 4; ++ni)
        acc[mi][ni] = wmma_bf16(af[mi], bfr[ni], acc[mi][ni]);

    // next tile staged + everyone done reading `cur` before it is reused
    wait_async0();
    __syncthreads();
  }

  // C/D layout: lane<16 -> row r, lane>=16 -> row r+8; col = lane&15
  (void)ldT;
#pragma unroll
  for (int mi = 0; mi < 4; ++mi)
#pragma unroll
    for (int ni = 0; ni < 4; ++ni)
#pragma unroll
      for (int r = 0; r < 8; ++r) {
        const int row = mbBlk + wm * 64 + mi * 16 + r + 8 * half;
        const int col = nbBlk + wn * 64 + ni * 16 + mrow;
        const float v = acc[mi][ni][r];
        if constexpr (MODE == 0)      ((__bf16*)out)[(size_t)row * N + col] = f2bf(v);
        else if constexpr (MODE == 1) ((__bf16*)out)[(size_t)col * ldT + row] = f2bf(v);
        else                          ((float*)out)[(size_t)row * N + col] = v;
      }
}

// ---------------------------------------------------------------------------
// Flash attention, causal. One wave per block; wave owns 16 query rows of one
// (batch, head). Iterates 32-key tiles up to the causal boundary.
//   Q,K : bf16 [4096, 768]   (row = b*2048+s, col = h*64+d)
//   Vt  : bf16 [768, 4096]   (row = h*64+d,  col = b*2048+s)
//   O   : bf16 [4096, 768]
__global__ __launch_bounds__(32)
void attn_flash(const __bf16* __restrict__ Q, const __bf16* __restrict__ Km,
                const __bf16* __restrict__ Vt, __bf16* __restrict__ O) {
  __shared__ __bf16 Plds[16 * 32];           // P tile staged C-layout -> A-layout
  const int lane = threadIdx.x;
  const int mrow = lane & 15, half = lane >> 4, h8 = half * 8;
  const int h = blockIdx.y, b = blockIdx.z;
  const int qb = blockIdx.x << 4;
  const int rowBase = b * SEQ;
  const int hc = h * DKH;

  // Q fragments for dk chunks [0,32) and [32,64) — loaded once
  const __bf16* qp = Q + (size_t)(rowBase + qb + mrow) * D_MODEL + hc;
  const bf16x16 aq0 = ldfrag(qp + h8,      qp + 16 + h8);
  const bf16x16 aq1 = ldfrag(qp + 32 + h8, qp + 48 + h8);

  float mrun[8], lrun[8];
#pragma unroll
  for (int r = 0; r < 8; ++r) { mrun[r] = -1e30f; lrun[r] = 0.f; }
  f32x8 od[4] = {};

  const int qEnd = qb + 15;
  for (int kb = 0; kb <= qEnd; kb += 32) {
    // ---- scores: 16q x 32k, contraction dk=64 -> 4 WMMAs ----
    f32x8 s0 = {}, s1 = {};
    const __bf16* kp0 = Km + (size_t)(rowBase + kb + mrow) * D_MODEL + hc;
    const __bf16* kp1 = kp0 + (size_t)16 * D_MODEL;
    s0 = wmma_bf16(aq0, ldfrag(kp0 + h8,      kp0 + 16 + h8), s0);
    s0 = wmma_bf16(aq1, ldfrag(kp0 + 32 + h8, kp0 + 48 + h8), s0);
    s1 = wmma_bf16(aq0, ldfrag(kp1 + h8,      kp1 + 16 + h8), s1);
    s1 = wmma_bf16(aq1, ldfrag(kp1 + 32 + h8, kp1 + 48 + h8), s1);

    // ---- causal mask + scale + online softmax (per-row stats replicated
    //      across each 16-lane half via shfl_xor reductions) ----
    float corr[8];
#pragma unroll
    for (int r = 0; r < 8; ++r) {
      const int q = qb + r + 8 * half;
      float v0 = s0[r] * 0.125f;                    // 1/sqrt(64)
      float v1 = s1[r] * 0.125f;
      if (kb + mrow      > q) v0 = -1e30f;
      if (kb + 16 + mrow > q) v1 = -1e30f;
      float rm = fmaxf(v0, v1);
      rm = fmaxf(rm, __shfl_xor(rm, 1, 32));
      rm = fmaxf(rm, __shfl_xor(rm, 2, 32));
      rm = fmaxf(rm, __shfl_xor(rm, 4, 32));
      rm = fmaxf(rm, __shfl_xor(rm, 8, 32));
      const float mnew = fmaxf(mrun[r], rm);
      const float c    = __expf(mrun[r] - mnew);
      const float e0   = __expf(v0 - mnew);
      const float e1   = __expf(v1 - mnew);
      float sum = e0 + e1;
      sum += __shfl_xor(sum, 1, 32);
      sum += __shfl_xor(sum, 2, 32);
      sum += __shfl_xor(sum, 4, 32);
      sum += __shfl_xor(sum, 8, 32);
      lrun[r] = lrun[r] * c + sum;
      mrun[r] = mnew;
      corr[r] = c;
      const int m = r + 8 * half;
      Plds[m * 32 + mrow]      = f2bf(e0);
      Plds[m * 32 + 16 + mrow] = f2bf(e1);
    }
    __syncthreads();  // single-wave WG: cheap; orders DS write->read
    const bf16x16 ap = ldfrag((const __bf16*)&Plds[mrow * 32 + h8],
                              (const __bf16*)&Plds[mrow * 32 + 16 + h8]);
    __syncthreads();

    // ---- O += P @ V  (contraction over 32 keys, 4 dk col-tiles) ----
#pragma unroll
    for (int d = 0; d < 4; ++d) {
      const __bf16* vp = Vt + (size_t)(hc + d * 16 + mrow) * TOKENS + rowBase + kb;
      const bf16x16 bv = ldfrag(vp + h8, vp + 16 + h8);
#pragma unroll
      for (int r = 0; r < 8; ++r) od[d][r] *= corr[r];
      od[d] = wmma_bf16(ap, bv, od[d]);
    }
  }

  // ---- normalize and store ----
#pragma unroll
  for (int r = 0; r < 8; ++r) {
    const float inv = 1.0f / lrun[r];
    const int row = rowBase + qb + r + 8 * half;
#pragma unroll
    for (int d = 0; d < 4; ++d)
      O[(size_t)row * D_MODEL + hc + d * 16 + mrow] = f2bf(od[d][r] * inv);
  }
}

// ---------------------------------------------------------------------------
extern "C" void kernel_launch(void* const* d_in, const int* in_sizes, int n_in,
                              void* d_out, int out_size, void* d_ws, size_t ws_size,
                              hipStream_t stream) {
  const float* x  = (const float*)d_in[0];
  const float* wq = (const float*)d_in[1];
  const float* wk = (const float*)d_in[2];
  const float* wv = (const float*)d_in[3];
  const float* wo = (const float*)d_in[4];

  // workspace layout (bytes), all 16B aligned
  char* ws = (char*)d_ws;
  const size_t XBF = TOKENS * (size_t)D_MODEL * 2;        // 6,291,456
  const size_t WBF = D_MODEL * (size_t)D_MODEL * 2;       // 1,179,648
  __bf16* xbf  = (__bf16*)(ws);
  __bf16* wqbf = (__bf16*)(ws + XBF);
  __bf16* wkbf = (__bf16*)(ws + XBF + WBF);
  __bf16* wvbf = (__bf16*)(ws + XBF + 2 * WBF);
  __bf16* wobf = (__bf16*)(ws + XBF + 3 * WBF);
  __bf16* Qbf  = (__bf16*)(ws + XBF + 4 * WBF);
  __bf16* Kbf  = (__bf16*)(ws + 2 * XBF + 4 * WBF);
  __bf16* Vt   = (__bf16*)(ws + 3 * XBF + 4 * WBF);       // [768, 4096] transposed
  __bf16* Obf  = (__bf16*)(ws + 4 * XBF + 4 * WBF);
  // total: 5*XBF + 4*WBF = 36,175,872 bytes

  const int nx = TOKENS * D_MODEL;
  const int nw = D_MODEL * D_MODEL;
  cvt_f32_bf16<<<(nx + 255) / 256, 256, 0, stream>>>(x,  xbf,  nx);
  cvt_f32_bf16<<<(nw + 255) / 256, 256, 0, stream>>>(wq, wqbf, nw);
  cvt_f32_bf16<<<(nw + 255) / 256, 256, 0, stream>>>(wk, wkbf, nw);
  cvt_f32_bf16<<<(nw + 255) / 256, 256, 0, stream>>>(wv, wvbf, nw);
  cvt_f32_bf16<<<(nw + 255) / 256, 256, 0, stream>>>(wo, wobf, nw);

  // (4096/128) * (768/256) = 32*3 = 96 blocks, each 128x256 C tile
  const int gemmBlocks = (TOKENS / 128) * (D_MODEL / 256);
  gemm_bf16_nt<0><<<gemmBlocks, 256, 0, stream>>>(xbf, wqbf, Qbf, TOKENS, D_MODEL, D_MODEL, 0);
  gemm_bf16_nt<0><<<gemmBlocks, 256, 0, stream>>>(xbf, wkbf, Kbf, TOKENS, D_MODEL, D_MODEL, 0);
  gemm_bf16_nt<1><<<gemmBlocks, 256, 0, stream>>>(xbf, wvbf, Vt,  TOKENS, D_MODEL, D_MODEL, TOKENS);

  attn_flash<<<dim3(SEQ / 16, NHEAD, BATCH), 32, 0, stream>>>(Qbf, Kbf, Vt, Obf);

  gemm_bf16_nt<2><<<gemmBlocks, 256, 0, stream>>>(Obf, wobf, d_out, TOKENS, D_MODEL, D_MODEL, 0);
}
